// Decoder_45475113730598
// MI455X (gfx1250) — compile-verified, matrix operand order checked by reading, not run on hardware
//
#include <hip/hip_runtime.h>

typedef __bf16 bf16;
typedef __attribute__((ext_vector_type(16))) __bf16 v16bf;
typedef __attribute__((ext_vector_type(8)))  __bf16 v8bf;
typedef __attribute__((ext_vector_type(8)))  float  v8f;
typedef __attribute__((ext_vector_type(4)))  float  v4f;
typedef __attribute__((ext_vector_type(4)))  int    v4i;

#define D_MODEL 1024
#define N_HEADS 16
#define DKH     64
#define D_FF    4096
#define SEQ     2048
#define BATCH   2
#define MROWS   (BATCH * SEQ)

// ---------------------------------------------------------------- CDNA5 async-to-LDS path (guarded)

#if defined(__AMDGCN__) && __has_builtin(__builtin_amdgcn_global_load_async_to_lds_b128)
#define HAVE_ASYNC_LDS 1
typedef __attribute__((address_space(1))) v4i gv4i;   // global (prints as __device__)
typedef __attribute__((address_space(3))) v4i lv4i;   // LDS    (prints as __shared__)
#else
#define HAVE_ASYNC_LDS 0
#endif

// ---------------------------------------------------------------- helpers

__device__ __forceinline__ bf16 f2bf(float f) {
  unsigned u = __builtin_bit_cast(unsigned, f);
  u += 0x7FFFu + ((u >> 16) & 1u);          // round-to-nearest-even
  unsigned short s = (unsigned short)(u >> 16);
  return __builtin_bit_cast(bf16, s);
}

__device__ __forceinline__ v8bf ld8(const bf16* p) { return *(const v8bf*)p; }

__device__ __forceinline__ v16bf mkfrag(v8bf lo, v8bf hi) {
  union { v16bf v; v8bf h[2]; } u;
  u.h[0] = lo; u.h[1] = hi;                 // elems 0..7 = K base, 8..15 = K base+16
  return u.v;
}

// 16-byte global->LDS copy: async DMA (ASYNCcnt) on CDNA5, plain copy otherwise
__device__ __forceinline__ void async_copy16(const bf16* g, bf16* l) {
#if HAVE_ASYNC_LDS
  __builtin_amdgcn_global_load_async_to_lds_b128((gv4i*)g, (lv4i*)l, 0, 0);
#else
  *(v8bf*)l = *(const v8bf*)g;
#endif
}

__device__ __forceinline__ void wait_async() {
#if HAVE_ASYNC_LDS
#if __has_builtin(__builtin_amdgcn_s_wait_asynccnt)
  __builtin_amdgcn_s_wait_asynccnt(0);
#else
  asm volatile("s_wait_asynccnt 0" ::: "memory");
#endif
#endif
}

// ---------------------------------------------------------------- convert f32 -> bf16

__global__ __launch_bounds__(256)
void convert_kernel(const float* __restrict__ in, bf16* __restrict__ out, int n) {
  int i = (blockIdx.x * 256 + threadIdx.x) * 4;
  if (i + 3 < n) {
    v4f t = *(const v4f*)(in + i);
    out[i + 0] = f2bf(t[0]);
    out[i + 1] = f2bf(t[1]);
    out[i + 2] = f2bf(t[2]);
    out[i + 3] = f2bf(t[3]);
  }
}

// ---------------------------------------------------------------- layernorm (Bessel std, eps on std)

__device__ __forceinline__ float block_sum(float v, float* smem) {
  v += __shfl_xor(v, 1, 32);  v += __shfl_xor(v, 2, 32);
  v += __shfl_xor(v, 4, 32);  v += __shfl_xor(v, 8, 32);
  v += __shfl_xor(v, 16, 32);
  const int w = threadIdx.x >> 5;
  if ((threadIdx.x & 31) == 0) smem[w] = v;
  __syncthreads();
  float r = 0.f;
#pragma unroll
  for (int i = 0; i < 8; ++i) r += smem[i];
  __syncthreads();
  return r;
}

__global__ __launch_bounds__(256)
void layernorm_kernel(const float* __restrict__ x, const float* __restrict__ g,
                      const float* __restrict__ b, bf16* __restrict__ out) {
  __shared__ float red[8];
  const int row = blockIdx.x;
  const float* xr = x + (size_t)row * D_MODEL;
  float v[4];
#pragma unroll
  for (int i = 0; i < 4; ++i) v[i] = xr[threadIdx.x + 256 * i];
  float s = block_sum(v[0] + v[1] + v[2] + v[3], red);
  float mean = s * (1.0f / D_MODEL);
  float d2 = 0.f;
#pragma unroll
  for (int i = 0; i < 4; ++i) { float d = v[i] - mean; d2 += d * d; }
  float ss = block_sum(d2, red);
  float stdv = sqrtf(ss * (1.0f / (D_MODEL - 1)));
  float inv = 1.0f / (stdv + 1e-6f);
#pragma unroll
  for (int i = 0; i < 4; ++i) {
    int c = threadIdx.x + 256 * i;
    out[(size_t)row * D_MODEL + c] = f2bf(g[c] * ((v[i] - mean) * inv) + b[c]);
  }
}

// ---------------------------------------------------------------- bf16 GEMM: out = act(A@W + bias [+ res])
// A: [M,K] bf16 row-major.  W: [K,N] f32 row-major (bf16-converted on tile stage).
// 256 thr = 8 waves, block tile 128x128, K step 32, wave tile 64x32 (4x2 wmma tiles).
// Double-buffered LDS; A tile staged via async-to-LDS, W tile converted post-compute.

template<bool RELU, typename OutT>
__global__ __launch_bounds__(256)
void gemm_bf16_kernel(const bf16* __restrict__ A, const float* __restrict__ W,
                      const float* __restrict__ bias, const float* __restrict__ res,
                      OutT* __restrict__ out, int Mdim, int Ndim, int Kdim) {
  __shared__ bf16 As[2][128][32];   // [buf][m][k]
  __shared__ bf16 Bt[2][128][32];   // [buf][n][k]  (transposed weight tile)

  const int tid  = threadIdx.x;
  const int lane = tid & 31;
  const int wave = tid >> 5;
  const int l16  = lane & 15;
  const int hs   = lane >> 4;           // half select
  const int m0   = blockIdx.y * 128;
  const int n0   = blockIdx.x * 128;
  const int wm   = (wave >> 2) * 64;    // 0 / 64
  const int wn   = (wave & 3)  * 32;    // 0..96

  v8f acc[4][2];
#pragma unroll
  for (int i = 0; i < 4; ++i)
#pragma unroll
    for (int j = 0; j < 2; ++j) acc[i][j] = (v8f){};

  const int arow = tid >> 1;            // 0..127
  const int ak   = (tid & 1) * 16;      // 0 / 16
  const int bkk  = tid >> 3;            // 0..31
  const int bn   = (tid & 7) * 16;      // 0..112

  const bf16*  aB = A + (size_t)(m0 + arow) * Kdim + ak;
  const float* wB = W + (size_t)bkk * Ndim + n0 + bn;

  // ---- stage tile 0
  {
    async_copy16(aB, &As[0][arow][ak]);
    async_copy16(aB + 8, &As[0][arow][ak + 8]);
    const float* wp = wB;
#pragma unroll
    for (int i = 0; i < 16; i += 4) {
      v4f wv = *(const v4f*)(wp + i);
#pragma unroll
      for (int j = 0; j < 4; ++j) Bt[0][bn + i + j][bkk] = f2bf(wv[j]);
    }
  }
  wait_async();
  __syncthreads();

  int cur = 0;
  for (int k0 = 0; k0 < Kdim; k0 += 32) {
    const int  nxt  = cur ^ 1;
    const bool more = (k0 + 32) < Kdim;

    // ---- issue next-tile traffic before compute (latency hidden behind WMMA)
    v4f w0 = {}, w1 = {}, w2 = {}, w3 = {};
    if (more) {
      const bf16* ap = aB + (size_t)(k0 + 32);
      async_copy16(ap, &As[nxt][arow][ak]);
      async_copy16(ap + 8, &As[nxt][arow][ak + 8]);
      const float* wp = wB + (size_t)(k0 + 32) * Ndim;
      w0 = *(const v4f*)(wp);
      w1 = *(const v4f*)(wp + 4);
      w2 = *(const v4f*)(wp + 8);
      w3 = *(const v4f*)(wp + 12);
      if (k0 + 64 < Kdim) {                                   // warm GL2 two steps out
        __builtin_prefetch(wB + (size_t)(k0 + 64) * Ndim, 0, 1);
        __builtin_prefetch(aB + (size_t)(k0 + 64), 0, 1);
      }
    }

    // ---- compute on current buffer
    v16bf aF[4], bF[2];
#pragma unroll
    for (int mt = 0; mt < 4; ++mt) {
      const bf16* p = &As[cur][wm + mt * 16 + l16][hs * 8];
      aF[mt] = mkfrag(ld8(p), ld8(p + 16));
    }
#pragma unroll
    for (int nt = 0; nt < 2; ++nt) {
      const bf16* p = &Bt[cur][wn + nt * 16 + l16][hs * 8];
      bF[nt] = mkfrag(ld8(p), ld8(p + 16));
    }
#pragma unroll
    for (int mt = 0; mt < 4; ++mt)
#pragma unroll
      for (int nt = 0; nt < 2; ++nt)
        acc[mt][nt] = __builtin_amdgcn_wmma_f32_16x16x32_bf16(
            false, aF[mt], false, bF[nt], (short)0, acc[mt][nt], false, false);

    // ---- finish next-tile weight transpose (f32->bf16) after the WMMAs
    if (more) {
      v4f wv[4] = {w0, w1, w2, w3};
#pragma unroll
      for (int i = 0; i < 16; i += 4)
#pragma unroll
        for (int j = 0; j < 4; ++j) Bt[nxt][bn + i + j][bkk] = f2bf(wv[i >> 2][j]);
    }
    wait_async();
    __syncthreads();
    cur = nxt;
  }

  // ---- epilogue: bias + optional residual + optional relu
#pragma unroll
  for (int mt = 0; mt < 4; ++mt) {
#pragma unroll
    for (int nt = 0; nt < 2; ++nt) {
      const int n = n0 + wn + nt * 16 + l16;
      const float bv = bias[n];
#pragma unroll
      for (int e = 0; e < 8; ++e) {
        const int m = m0 + wm + mt * 16 + e + hs * 8;
        float v = acc[mt][nt][e] + bv;
        if (res) v += res[(size_t)m * Ndim + n];
        if (RELU) v = fmaxf(v, 0.f);
        if constexpr (sizeof(OutT) == 2)
          ((bf16*)out)[(size_t)m * Ndim + n] = f2bf(v);
        else
          ((float*)out)[(size_t)m * Ndim + n] = v;
      }
    }
  }
}

// ---------------------------------------------------------------- flash attention
// 128 thr = 4 waves; block = 64 queries (16/wave), KV chunk = 32, dk = 64.
// grid: (SEQ/64, N_HEADS, BATCH)

__global__ __launch_bounds__(128)
void attention_kernel(const bf16* __restrict__ Q, const bf16* __restrict__ K,
                      const bf16* __restrict__ V, bf16* __restrict__ O, int causal) {
  __shared__ bf16 Ks[32][64];        // [j][d]
  __shared__ bf16 Vt[64][32];        // [d][j]  (transposed)
  __shared__ bf16 Ps[4][16][32];     // per-wave P relayout patch

  const int tid  = threadIdx.x;
  const int lane = tid & 31;
  const int w    = tid >> 5;
  const int l16  = lane & 15;
  const int hs   = lane >> 4;
  const int qt   = blockIdx.x;
  const int h    = blockIdx.y;
  const size_t rowbase = (size_t)blockIdx.z * SEQ;
  const int hd   = h * DKH;
  const int qrow = qt * 64 + w * 16;

  // Q fragments (A layout): row = l16, two 32-wide d chunks
  v16bf qF[2];
  {
    const bf16* qp = Q + (rowbase + qrow + l16) * D_MODEL + hd;
#pragma unroll
    for (int t = 0; t < 2; ++t) {
      const bf16* p = qp + t * 32 + hs * 8;
      qF[t] = mkfrag(ld8(p), ld8(p + 16));
    }
  }

  v8f o[4];
#pragma unroll
  for (int t = 0; t < 4; ++t) o[t] = (v8f){};
  float mrow[8], lrow[8];
#pragma unroll
  for (int e = 0; e < 8; ++e) { mrow[e] = -3.0e38f; lrow[e] = 0.f; }

  const int jend = causal ? (qt * 64 + 64) : SEQ;
  const int cj = tid >> 2;            // 0..31
  const int cd = (tid & 3) * 16;      // 0..48

  for (int jc = 0; jc < jend; jc += 32) {
    // stage K chunk row-major via async DMA
    const bf16* kp = K + (rowbase + jc + cj) * D_MODEL + hd + cd;
    async_copy16(kp, &Ks[cj][cd]);
    async_copy16(kp + 8, &Ks[cj][cd + 8]);
    // stage V chunk transposed (manual: transpose can't be DMA'd)
    const bf16* vp = V + (rowbase + jc + cj) * D_MODEL + hd + cd;
    v8bf v0 = ld8(vp), v1 = ld8(vp + 8);
#pragma unroll
    for (int i = 0; i < 8; ++i) { Vt[cd + i][cj] = v0[i]; Vt[cd + 8 + i][cj] = v1[i]; }
    wait_async();
    __syncthreads();

    // scores = Q K^T : two 16-col subtiles, K-dim 64 -> 2 wmma each
    v8f sc[2];
#pragma unroll
    for (int s2 = 0; s2 < 2; ++s2) {
      v8f a = (v8f){};
#pragma unroll
      for (int t = 0; t < 2; ++t) {
        const bf16* p = &Ks[s2 * 16 + l16][t * 32 + hs * 8];
        v16bf bK = mkfrag(ld8(p), ld8(p + 16));
        a = __builtin_amdgcn_wmma_f32_16x16x32_bf16(false, qF[t], false, bK,
                                                    (short)0, a, false, false);
      }
      sc[s2] = a;
    }

    // scale + mask + online softmax (rows live in a 16-lane half, elem e)
    const float scale = 0.125f;       // 1/sqrt(64)
    float p0a[8], p1a[8], cm[8];
#pragma unroll
    for (int e = 0; e < 8; ++e) {
      const int q = qrow + hs * 8 + e;
      float s0 = sc[0][e] * scale;
      float s1 = sc[1][e] * scale;
      if (causal) {
        if (jc + l16 > q)      s0 = -1e9f;
        if (jc + 16 + l16 > q) s1 = -1e9f;
      }
      p0a[e] = s0; p1a[e] = s1;
      float m = fmaxf(s0, s1);
      m = fmaxf(m, __shfl_xor(m, 1, 32));
      m = fmaxf(m, __shfl_xor(m, 2, 32));
      m = fmaxf(m, __shfl_xor(m, 4, 32));
      m = fmaxf(m, __shfl_xor(m, 8, 32));
      cm[e] = m;
    }
#pragma unroll
    for (int e = 0; e < 8; ++e) {
      float mn = fmaxf(mrow[e], cm[e]);
      float sf = __expf(mrow[e] - mn);
      float p0 = __expf(p0a[e] - mn);
      float p1 = __expf(p1a[e] - mn);
      float rs = p0 + p1;
      rs += __shfl_xor(rs, 1, 32);
      rs += __shfl_xor(rs, 2, 32);
      rs += __shfl_xor(rs, 4, 32);
      rs += __shfl_xor(rs, 8, 32);
      lrow[e] = lrow[e] * sf + rs;
      mrow[e] = mn;
#pragma unroll
      for (int t = 0; t < 4; ++t) o[t][e] *= sf;
      Ps[w][hs * 8 + e][l16]      = f2bf(p0);
      Ps[w][hs * 8 + e][16 + l16] = f2bf(p1);
    }

    // out += P V  (A = P 16x32, B = Vt chunks; wave-private Ps, same-wave LDS is in-order)
    {
      const bf16* pp = &Ps[w][l16][hs * 8];
      v16bf aP = mkfrag(ld8(pp), ld8(pp + 16));
#pragma unroll
      for (int t = 0; t < 4; ++t) {
        const bf16* vq = &Vt[t * 16 + l16][hs * 8];
        v16bf bV = mkfrag(ld8(vq), ld8(vq + 16));
        o[t] = __builtin_amdgcn_wmma_f32_16x16x32_bf16(false, aP, false, bV,
                                                       (short)0, o[t], false, false);
      }
    }
    __syncthreads();
  }

  // normalize + store
#pragma unroll
  for (int e = 0; e < 8; ++e) {
    float inv = 1.0f / lrow[e];
    const size_t m = rowbase + qrow + hs * 8 + e;
#pragma unroll
    for (int t = 0; t < 4; ++t)
      O[m * D_MODEL + hd + t * 16 + l16] = f2bf(o[t][e] * inv);
  }
}

// ---------------------------------------------------------------- launch

extern "C" void kernel_launch(void* const* d_in, const int* in_sizes, int n_in,
                              void* d_out, int out_size, void* d_ws, size_t ws_size,
                              hipStream_t stream) {
  (void)in_sizes; (void)n_in; (void)out_size; (void)ws_size;
  const float* x     = (const float*)d_in[0];
  const float* enc   = (const float*)d_in[1];
  // d_in[2] src_mask (all ones), d_in[3] tgt_mask (tril): handled analytically
  const float* sa_wq = (const float*)d_in[4];  const float* sa_bq = (const float*)d_in[5];
  const float* sa_wk = (const float*)d_in[6];  const float* sa_bk = (const float*)d_in[7];
  const float* sa_wv = (const float*)d_in[8];  const float* sa_bv = (const float*)d_in[9];
  const float* sa_wo = (const float*)d_in[10]; const float* sa_bo = (const float*)d_in[11];
  const float* ca_wq = (const float*)d_in[12]; const float* ca_bq = (const float*)d_in[13];
  const float* ca_wk = (const float*)d_in[14]; const float* ca_bk = (const float*)d_in[15];
  const float* ca_wv = (const float*)d_in[16]; const float* ca_bv = (const float*)d_in[17];
  const float* ca_wo = (const float*)d_in[18]; const float* ca_bo = (const float*)d_in[19];
  const float* ff_w1 = (const float*)d_in[20]; const float* ff_b1 = (const float*)d_in[21];
  const float* ff_w2 = (const float*)d_in[22]; const float* ff_b2 = (const float*)d_in[23];
  const float* n1_g  = (const float*)d_in[24]; const float* n1_b  = (const float*)d_in[25];
  const float* n2_g  = (const float*)d_in[26]; const float* n2_b  = (const float*)d_in[27];
  const float* n3_g  = (const float*)d_in[28]; const float* n3_b  = (const float*)d_in[29];

  char* ws = (char*)d_ws;
  const size_t MB = (size_t)1 << 20;
  bf16*  hbuf  = (bf16*)(ws + 0 * MB);    // 8 MB  LN output
  bf16*  ebuf  = (bf16*)(ws + 8 * MB);    // 8 MB  encoder bf16
  bf16*  qbuf  = (bf16*)(ws + 16 * MB);   // 8 MB
  bf16*  kbuf  = (bf16*)(ws + 24 * MB);   // 8 MB
  bf16*  vbuf  = (bf16*)(ws + 32 * MB);   // 8 MB
  bf16*  abuf  = (bf16*)(ws + 40 * MB);   // 8 MB  attention out
  float* x1    = (float*)(ws + 48 * MB);  // 16 MB residual stream 1
  float* x2    = (float*)(ws + 64 * MB);  // 16 MB residual stream 2
  bf16*  ffbuf = (bf16*)(ws + 16 * MB);   // 32 MB, overlays q/k/v/a (retired by then)

  dim3 bLN(256), gLN(MROWS);
  dim3 bG(256);
  dim3 gG(D_MODEL / 128, MROWS / 128);    // 8 x 32
  dim3 gGff(D_FF / 128, MROWS / 128);     // 32 x 32
  dim3 bA(128), gA(SEQ / 64, N_HEADS, BATCH);

  // encoder output -> bf16 (used by cross-attn K/V projections)
  convert_kernel<<<(MROWS * D_MODEL) / 1024, 256, 0, stream>>>(enc, ebuf, MROWS * D_MODEL);

  // ---- block 1: self attention
  layernorm_kernel<<<gLN, bLN, 0, stream>>>(x, n1_g, n1_b, hbuf);
  gemm_bf16_kernel<false, bf16><<<gG, bG, 0, stream>>>(hbuf, sa_wq, sa_bq, nullptr, qbuf, MROWS, D_MODEL, D_MODEL);
  gemm_bf16_kernel<false, bf16><<<gG, bG, 0, stream>>>(hbuf, sa_wk, sa_bk, nullptr, kbuf, MROWS, D_MODEL, D_MODEL);
  gemm_bf16_kernel<false, bf16><<<gG, bG, 0, stream>>>(hbuf, sa_wv, sa_bv, nullptr, vbuf, MROWS, D_MODEL, D_MODEL);
  attention_kernel<<<gA, bA, 0, stream>>>(qbuf, kbuf, vbuf, abuf, 1);
  gemm_bf16_kernel<false, float><<<gG, bG, 0, stream>>>(abuf, sa_wo, sa_bo, x, x1, MROWS, D_MODEL, D_MODEL);

  // ---- block 2: cross attention
  layernorm_kernel<<<gLN, bLN, 0, stream>>>(x1, n2_g, n2_b, hbuf);
  gemm_bf16_kernel<false, bf16><<<gG, bG, 0, stream>>>(hbuf, ca_wq, ca_bq, nullptr, qbuf, MROWS, D_MODEL, D_MODEL);
  gemm_bf16_kernel<false, bf16><<<gG, bG, 0, stream>>>(ebuf, ca_wk, ca_bk, nullptr, kbuf, MROWS, D_MODEL, D_MODEL);
  gemm_bf16_kernel<false, bf16><<<gG, bG, 0, stream>>>(ebuf, ca_wv, ca_bv, nullptr, vbuf, MROWS, D_MODEL, D_MODEL);
  attention_kernel<<<gA, bA, 0, stream>>>(qbuf, kbuf, vbuf, abuf, 0);
  gemm_bf16_kernel<false, float><<<gG, bG, 0, stream>>>(abuf, ca_wo, ca_bo, x1, x2, MROWS, D_MODEL, D_MODEL);

  // ---- block 3: FFN
  layernorm_kernel<<<gLN, bLN, 0, stream>>>(x2, n3_g, n3_b, hbuf);
  gemm_bf16_kernel<true, bf16><<<gGff, bG, 0, stream>>>(hbuf, ff_w1, ff_b1, nullptr, ffbuf, MROWS, D_FF, D_MODEL);
  gemm_bf16_kernel<false, float><<<gG, bG, 0, stream>>>(ffbuf, ff_w2, ff_b2, x2, (float*)d_out, MROWS, D_MODEL, D_FF);
}
